// PartialFC_V2_73065983640000
// MI455X (gfx1250) — matrix operand order repaired and checked by reading, not compiled, not run on hardware
//
#include <hip/hip_runtime.h>
#include <math.h>

#define BATCH 1024
#define EMBD  512
#define NCLS  100000
#define CBLK  64            // classes per workgroup
#define SCALE_S 64.0f
#define MARGIN_M 0.5f
#define PI_F 3.14159265358979323846f

typedef __attribute__((ext_vector_type(16))) __bf16 v16bf;
typedef __attribute__((ext_vector_type(8)))  float  v8f;

union Frag16 {
    v16bf v;
    uint4 q[2];
};

__device__ __forceinline__ unsigned short f32_to_bf16_rne(float x) {
    unsigned int u = __float_as_uint(x);
    unsigned int r = u + 0x7FFFu + ((u >> 16) & 1u);
    return (unsigned short)(r >> 16);
}
__device__ __forceinline__ float bf16_bits_to_f32(unsigned short h) {
    return __uint_as_float(((unsigned int)h) << 16);
}

// ---------------------------------------------------------------------------
// Kernel 1: normalize each embedding row, split into bf16 hi/lo in workspace.
// Also zeros the per-row softmax accumulators (harness does not re-zero ws).
// ---------------------------------------------------------------------------
__global__ __launch_bounds__(128) void embed_prep_kernel(
    const float* __restrict__ emb, unsigned short* __restrict__ a_hi,
    unsigned short* __restrict__ a_lo, float* __restrict__ rowsum,
    float* __restrict__ target) {
    const int b = blockIdx.x;
    const int tid = threadIdx.x;
    const float* row = emb + (size_t)b * EMBD;
    float s = 0.f;
    for (int i = tid; i < EMBD; i += 128) { float v = row[i]; s += v * v; }
    s += __shfl_xor(s, 1);  s += __shfl_xor(s, 2);
    s += __shfl_xor(s, 4);  s += __shfl_xor(s, 8);
    s += __shfl_xor(s, 16);
    __shared__ float red[4];
    if ((tid & 31) == 0) red[tid >> 5] = s;
    __syncthreads();
    __shared__ float invn_sh;
    if (tid == 0) {
        invn_sh = rsqrtf(red[0] + red[1] + red[2] + red[3]);
        rowsum[b] = 0.f;
        target[b] = 0.f;
    }
    __syncthreads();
    const float invn = invn_sh;
    const int i4 = tid * 4;                       // one float4 per thread
    const float4 s4 = *(const float4*)(row + i4);
    const float xs[4] = {s4.x * invn, s4.y * invn, s4.z * invn, s4.w * invn};
    unsigned int h[4], l[4];
    #pragma unroll
    for (int j = 0; j < 4; ++j) {
        unsigned short hb = f32_to_bf16_rne(xs[j]);
        h[j] = hb;
        l[j] = f32_to_bf16_rne(xs[j] - bf16_bits_to_f32(hb));
    }
    *(uint2*)(&a_hi[(size_t)b * EMBD + i4]) = make_uint2(h[0] | (h[1] << 16), h[2] | (h[3] << 16));
    *(uint2*)(&a_lo[(size_t)b * EMBD + i4]) = make_uint2(l[0] | (l[1] << 16), l[2] | (l[3] << 16));
}

// ---------------------------------------------------------------------------
// Kernel 2: fused GEMM (split-bf16 WMMA) + weight norms + ArcFace + sum(exp).
//   One WG per 64-class block: stage+convert the whole W block (64x512 hi/lo
//   = 128 KB LDS) ONCE, fuse ||w||^2 via ds_add_f32, then sweep all 8 batch
//   tiles against it. W leaves HBM exactly once (205 MB ~ 9us at 23.3 TB/s).
//   256 threads = 8 wave32; wave (wm,wn) owns 2x2 16x16 tiles per batch tile.
// ---------------------------------------------------------------------------
__global__ __launch_bounds__(256) void arcface_fused_kernel(
    const unsigned short* __restrict__ a_hi, const unsigned short* __restrict__ a_lo,
    const float* __restrict__ wgt, const int* __restrict__ labels,
    float* __restrict__ rowsum, float* __restrict__ target) {
    extern __shared__ __align__(16) unsigned short dynsmem[];
    unsigned short* sBhi = dynsmem;                    // 64*512
    unsigned short* sBlo = sBhi + CBLK * EMBD;         // 64*512
    unsigned short* sAhi = sBlo + CBLK * EMBD;         // 128*32
    unsigned short* sAlo = sAhi + 128 * 32;            // 128*32
    __shared__ float sq[CBLK];
    __shared__ float invw_sh[CBLK];

    const int tid    = threadIdx.x;
    const int lane   = tid & 31;
    const int wave   = tid >> 5;
    const int wave_m = wave & 3;   // 4 waves along M: 32 rows each
    const int wave_n = wave >> 2;  // 2 waves along N: 32 cols each
    const int khalf  = lane >> 4;  // lane half selects K sub-range (ISA layout)
    const int l16    = lane & 15;
    const int cls_base = blockIdx.x * CBLK;

    if (tid < CBLK) sq[tid] = 0.f;
    __syncthreads();

    // ---- stage + convert entire B block; fuse ||w||^2 accumulation ----
    #pragma unroll 4
    for (int i = 0; i < 32; ++i) {
        const int f   = tid + 256 * i;     // 0..8191 float4 slots
        const int row = f >> 7;            // 0..63
        const int kq  = f & 127;           // float4 within 512-wide row
        const int c   = cls_base + row;
        float4 s4 = make_float4(0.f, 0.f, 0.f, 0.f);
        if (c < NCLS) {
            const float* p = wgt + (size_t)c * EMBD + kq * 4;
            s4 = *(const float4*)p;
        }
        // next-iteration streaming prefetch
        {
            const int f2 = f + 256;
            const int c2 = cls_base + (f2 >> 7);
            if (i < 31 && c2 < NCLS)
                __builtin_prefetch(wgt + (size_t)c2 * EMBD + (f2 & 127) * 4, 0, 0);
        }
        atomicAdd(&sq[row], s4.x * s4.x + s4.y * s4.y + s4.z * s4.z + s4.w * s4.w);
        const float xs[4] = {s4.x, s4.y, s4.z, s4.w};
        unsigned int h[4], l[4];
        #pragma unroll
        for (int j = 0; j < 4; ++j) {
            unsigned short hb = f32_to_bf16_rne(xs[j]);
            h[j] = hb;
            l[j] = f32_to_bf16_rne(xs[j] - bf16_bits_to_f32(hb));
        }
        *(uint2*)(&sBhi[row * EMBD + kq * 4]) = make_uint2(h[0] | (h[1] << 16), h[2] | (h[3] << 16));
        *(uint2*)(&sBlo[row * EMBD + kq * 4]) = make_uint2(l[0] | (l[1] << 16), l[2] | (l[3] << 16));
    }
    __syncthreads();
    if (tid < CBLK)
        invw_sh[tid] = (cls_base + tid < NCLS) ? rsqrtf(sq[tid]) : 0.f;
    // (next barrier below covers this write before any read)

    // per-lane class metadata for the epilogue
    float invw[2];
    int   cls[2];
    bool  cvalid[2];

    // ---- sweep the 8 batch tiles over the resident W block ----
    for (int bt = 0; bt < 8; ++bt) {
        const int row_base = bt * 128;
        const v8f vzero = {0.f, 0.f, 0.f, 0.f, 0.f, 0.f, 0.f, 0.f};
        v8f acc[2][2];
        acc[0][0] = vzero; acc[0][1] = vzero; acc[1][0] = vzero; acc[1][1] = vzero;

        for (int kc = 0; kc < EMBD / 32; ++kc) {
            __syncthreads();   // protect sA from overwrite (and invw_sh write)

            // stage A tile (128x32 hi/lo): straight bf16 copy, 512 uint4/array
            #pragma unroll
            for (int i = 0; i < 2; ++i) {
                const int f   = tid + 256 * i;  // 0..511 uint4 slots
                const int row = f >> 2;         // 0..127
                const int kq  = f & 3;          // uint4 (8 ushort) within 32-k
                const size_t g = (size_t)(row_base + row) * EMBD + kc * 32 + kq * 8;
                *(uint4*)(&sAhi[row * 32 + kq * 8]) = *(const uint4*)(&a_hi[g]);
                *(uint4*)(&sAlo[row * 32 + kq * 8]) = *(const uint4*)(&a_lo[g]);
            }
            __syncthreads();

            // fragments per ISA 16-bit A(16x32)/B(32x16) lane layouts
            Frag16 ah[2], al[2], bh[2], bl[2];
            #pragma unroll
            for (int mt = 0; mt < 2; ++mt) {
                const int m = wave_m * 32 + mt * 16 + l16;
                const unsigned short* ph = &sAhi[m * 32 + khalf * 8];
                const unsigned short* pl = &sAlo[m * 32 + khalf * 8];
                ah[mt].q[0] = *(const uint4*)(ph);
                ah[mt].q[1] = *(const uint4*)(ph + 16);
                al[mt].q[0] = *(const uint4*)(pl);
                al[mt].q[1] = *(const uint4*)(pl + 16);
            }
            #pragma unroll
            for (int nt = 0; nt < 2; ++nt) {
                const int n = wave_n * 32 + nt * 16 + l16;
                const unsigned short* ph = &sBhi[n * EMBD + kc * 32 + khalf * 8];
                const unsigned short* pl = &sBlo[n * EMBD + kc * 32 + khalf * 8];
                bh[nt].q[0] = *(const uint4*)(ph);
                bh[nt].q[1] = *(const uint4*)(ph + 16);
                bl[nt].q[0] = *(const uint4*)(pl);
                bl[nt].q[1] = *(const uint4*)(pl + 16);
            }

            // split-bf16 fp32 emulation: ah*bh + ah*bl + al*bh
            #pragma unroll
            for (int mt = 0; mt < 2; ++mt) {
                #pragma unroll
                for (int nt = 0; nt < 2; ++nt) {
                    acc[mt][nt] = __builtin_amdgcn_wmma_f32_16x16x32_bf16(
                        false, ah[mt].v, false, bh[nt].v, (short)0, acc[mt][nt], false, false);
                    acc[mt][nt] = __builtin_amdgcn_wmma_f32_16x16x32_bf16(
                        false, ah[mt].v, false, bl[nt].v, (short)0, acc[mt][nt], false, false);
                    acc[mt][nt] = __builtin_amdgcn_wmma_f32_16x16x32_bf16(
                        false, al[mt].v, false, bh[nt].v, (short)0, acc[mt][nt], false, false);
                }
            }
        }

        // ---- epilogue for this batch tile ----
        #pragma unroll
        for (int nt = 0; nt < 2; ++nt) {
            const int cl = wave_n * 32 + nt * 16 + l16;
            cls[nt]    = cls_base + cl;
            cvalid[nt] = (cls[nt] < NCLS);
            invw[nt]   = invw_sh[cl];
        }
        #pragma unroll
        for (int mt = 0; mt < 2; ++mt) {
            #pragma unroll
            for (int r = 0; r < 8; ++r) {
                // C/D layout: VGPR r -> M=r (lanes 0-15) / M=r+8 (lanes 16-31)
                const int b   = row_base + wave_m * 32 + mt * 16 + khalf * 8 + r;
                const int lbl = labels[b];
                float ts = 0.f;
                #pragma unroll
                for (int nt = 0; nt < 2; ++nt) {
                    float cosv = acc[mt][nt][r] * invw[nt];
                    cosv = fminf(fmaxf(cosv, -1.f), 1.f);
                    float t = 0.f;
                    if (cvalid[nt]) {
                        if (cls[nt] == lbl) {
                            float th = acosf(cosv) + MARGIN_M;
                            th = fminf(th, PI_F);
                            const float nl = SCALE_S * cosf(th);
                            target[b] = nl;          // unique writer per b
                            t = __expf(nl);
                        } else {
                            t = __expf(SCALE_S * cosv);
                        }
                    }
                    ts += t;
                }
                // sum over the 32 classes (2 per lane) held by this lane half
                ts += __shfl_xor(ts, 1);
                ts += __shfl_xor(ts, 2);
                ts += __shfl_xor(ts, 4);
                ts += __shfl_xor(ts, 8);
                if (l16 == 0) atomicAdd(&rowsum[b], ts);
            }
        }
    }
}

// ---------------------------------------------------------------------------
// Kernel 3: loss = -mean(log(max(exp(target - logsumexp), 1e-30)))
// ---------------------------------------------------------------------------
__global__ __launch_bounds__(256) void finalize_kernel(
    const float* __restrict__ rowsum, const float* __restrict__ target,
    float* __restrict__ out) {
    const int tid = threadIdx.x;
    float s = 0.f;
    for (int b = tid; b < BATCH; b += 256) {
        const float lse = logf(rowsum[b]);
        const float lp  = target[b] - lse;          // log p_target
        const float p   = __expf(lp);
        s += -logf(fmaxf(p, 1e-30f));
    }
    s += __shfl_xor(s, 1);  s += __shfl_xor(s, 2);
    s += __shfl_xor(s, 4);  s += __shfl_xor(s, 8);
    s += __shfl_xor(s, 16);
    __shared__ float red[8];
    if ((tid & 31) == 0) red[tid >> 5] = s;
    __syncthreads();
    if (tid == 0) {
        float t = 0.f;
        #pragma unroll
        for (int i = 0; i < 8; ++i) t += red[i];
        out[0] = t / (float)BATCH;
    }
}

// ---------------------------------------------------------------------------
extern "C" void kernel_launch(void* const* d_in, const int* in_sizes, int n_in,
                              void* d_out, int out_size, void* d_ws, size_t ws_size,
                              hipStream_t stream) {
    const float* emb    = (const float*)d_in[0];   // [1024, 512] f32
    const float* wgt    = (const float*)d_in[1];   // [100000, 512] f32
    const int*   labels = (const int*)d_in[2];     // [1024]
    float*       out    = (float*)d_out;           // scalar loss

    float* ws     = (float*)d_ws;
    float* rowsum = ws;                            // 1024 f32
    float* target = rowsum + BATCH;                // 1024 f32
    unsigned short* a_hi = (unsigned short*)(target + BATCH);   // 1024*512 bf16
    unsigned short* a_lo = a_hi + (size_t)BATCH * EMBD;         // 1024*512 bf16

    embed_prep_kernel<<<BATCH, 128, 0, stream>>>(emb, a_hi, a_lo, rowsum, target);

    const int nblocks = (NCLS + CBLK - 1) / CBLK;  // 1563
    const size_t lds_bytes = (size_t)(2 * CBLK * EMBD + 2 * 128 * 32) * sizeof(unsigned short);
    arcface_fused_kernel<<<nblocks, 256, lds_bytes, stream>>>(a_hi, a_lo, wgt, labels,
                                                              rowsum, target);
    finalize_kernel<<<1, 256, 0, stream>>>(rowsum, target, out);
}
